// PLSTMModel_65601330479668
// MI455X (gfx1250) — matrix operand order repaired and checked by reading, not compiled, not run on hardware
//
#include <hip/hip_runtime.h>
#include <stdint.h>

#define P_  4
#define FP_ 128
#define H_  512
#define C_  60
#define B_  256
#define T_  256
#define G3H   (3 * H_)        // 1536
#define NGATES (P_ * G3H)     // 6144 gate columns (p-major)
#define NTOT  (NGATES + H_)   // 6656 = gates + output-gate columns

typedef __attribute__((ext_vector_type(16))) __bf16 v16bf;
typedef __attribute__((ext_vector_type(8)))  float  v8f;
typedef __attribute__((ext_vector_type(2)))  float  v2f;
typedef __attribute__((ext_vector_type(2)))  __bf16 v2bf;

union Frag {
  v16bf   v;
  uint32_t u[8];
  uint4    q[2];
};

// float2 -> packed bf16x2 (RTNE): lowers to a single v_cvt_pk_bf16_f32.
static __device__ __forceinline__ uint32_t pack2(float a, float b) {
  v2f f; f.x = a; f.y = b;
  union { v2bf h; uint32_t u; } r;
  r.h = __builtin_convertvector(f, v2bf);
  return r.u;
}
static __device__ __forceinline__ uint16_t f2bf(float f) {
  union { __bf16 h; uint16_t u; } r;
  r.h = (__bf16)f;
  return r.u;
}
static __device__ __forceinline__ float sigmoidf_(float x) {
  return 1.0f / (1.0f + expf(-x));
}

// ---------------------------------------------------------------------------
// Weight pre-swizzle: fp32 [K,N] (split as srcA rows [0,Ka) + srcB rows [Ka,K))
// -> bf16 fragment-major layout:
//   half index = ((nt*nKC + kc)*32 + lane)*16 + e,  K = kc*32+8*(lane>>4)+(e&7)+16*(e>>3), N = nt*16+(lane&15)
// Matches the wave32 WMMA B-matrix layout (lane%16 = N col, 16 halves = K), so
// the GEMM kernel reads each B fragment as 32 contiguous bytes per lane.
// ---------------------------------------------------------------------------
__global__ void convert_w(const float* __restrict__ srcA, int Ka,
                          const float* __restrict__ srcB, int N, int nKC,
                          uint16_t* __restrict__ dst) {
  size_t o = (size_t)blockIdx.x * 256 + threadIdx.x;
  int e  = (int)(o & 15);
  int L  = (int)((o >> 4) & 31);
  size_t r = o >> 9;
  int kc = (int)(r % (size_t)nKC);
  int nt = (int)(r / (size_t)nKC);
  int K  = kc * 32 + ((L >> 4) << 3) + (e & 7) + ((e >> 3) << 4);
  int n  = nt * 16 + (L & 15);
  float f = (K < Ka) ? srcA[(size_t)K * N + n]
                     : srcB[(size_t)(K - Ka) * N + n];
  dst[o] = f2bf(f);
}

// ---------------------------------------------------------------------------
// Per-timestep pre-activation GEMM for one cell:
//   columns [0, 6144):  gates[b, p, n] = [x_p | h] @ [Wx_p ; Wh_p] + b[p,n]
//   columns [6144,6656): ogate[b, n]   = [x_flat | h] @ [Wxo ; Who] + bo[n]
// Block: 256 threads (8 waves), 64(M) x 128(N) tile; wave = 16x64 strip.
// Double-buffered LDS A staging: one barrier per k-chunk; next chunk's fp32
// loads are issued right after the barrier and fly during the WMMA block.
// ---------------------------------------------------------------------------
__global__ __launch_bounds__(256, 1)
void gates_kernel(const float* __restrict__ x, int t,
                  const float* __restrict__ h,
                  const uint16_t* __restrict__ Wg,   // [4][20*96*512] halves
                  const uint16_t* __restrict__ Wo,   // [32*32*512] halves
                  const float* __restrict__ bg,      // [P, 3H]
                  const float* __restrict__ bo,      // [H]
                  float* __restrict__ gout)          // [B, NTOT]
{
  __shared__ uint32_t smemA[2][64 * 17]; // 2 x (64 rows x 32 halves), pitch 17

  const int tid    = threadIdx.x;
  const int lane   = tid & 31;
  const int wid    = tid >> 5;           // 0..7
  const int mw     = wid & 3;            // M sub-tile (16 rows each)
  const int nw     = wid >> 2;           // 0..1 (64 N cols each)
  const bool laneHi = (lane >= 16);

  const int  nt128 = blockIdx.x;         // 0..51
  const int  mtile = blockIdx.y * 64;
  const bool ogate = (nt128 >= 48);

  int p = 0, ncol0, nKC;
  const uint16_t* W;
  if (!ogate) {
    p     = nt128 / 12;
    ncol0 = (nt128 % 12) * 128;
    nKC   = (FP_ + H_) / 32;                         // 20 (K = 640)
    W     = Wg + (size_t)p * (20u * 96u * 512u);
  } else {
    ncol0 = (nt128 - 48) * 128;
    nKC   = (P_ * FP_ + H_) / 32;                    // 32 (K = 1024)
    W     = Wo;
  }

  v8f acc0 = {}; v8f acc1 = {}; v8f acc2 = {}; v8f acc3 = {};

  // A-staging coordinates (fixed per thread)
  const int sr     = tid >> 2;           // row within tile: 0..63
  const int kq     = (tid & 3) * 8;      // k-offset within 32-chunk: 0,8,16,24
  const int brow_s = mtile + sr;

  const int ntbase = (ncol0 >> 4) + nw * 4;
  const uint4* Wq  = (const uint4*)W;

  // per-thread global source address for A chunk kc
  auto srcp = [&](int kcx) -> const float* {
    const int kg = kcx * 32 + kq;
    if (!ogate) {
      if (kg < FP_) return x + ((((size_t)p * B_ + brow_s) * T_ + t) * FP_ + kg);
      return h + ((size_t)brow_s * H_ + (kg - FP_));
    } else {
      if (kg < P_ * FP_) {
        int pp = kg >> 7, f = kg & 127;
        return x + ((((size_t)pp * B_ + brow_s) * T_ + t) * FP_ + f);
      }
      return h + ((size_t)brow_s * H_ + (kg - P_ * FP_));
    }
  };

  // preload chunk 0 into registers
  float4 f0, f1;
  {
    const float* s = srcp(0);
    f0 = ((const float4*)s)[0];
    f1 = ((const float4*)s)[1];
  }

  for (int kc = 0; kc < nKC; ++kc) {
    // ---- commit staged chunk kc (bf16) into LDS buffer kc&1 ----
    uint32_t* dst = &smemA[kc & 1][sr * 17 + (kq >> 1)];
    dst[0] = pack2(f0.x, f0.y);
    dst[1] = pack2(f0.z, f0.w);
    dst[2] = pack2(f1.x, f1.y);
    dst[3] = pack2(f1.z, f1.w);
    __syncthreads();

    // ---- issue next chunk's fp32 loads (consumed next iteration) ----
    if (kc + 1 < nKC) {
      const float* s = srcp(kc + 1);
      f0 = ((const float4*)s)[0];
      f1 = ((const float4*)s)[1];
    }

    // ---- A fragment gather (16-bit A 16x32 layout; conflict-free pitch) ----
    Frag a;
    {
      const int m  = mw * 16 + (lane & 15);
      const int kb = laneHi ? 4 : 0;
      const uint32_t* row = &smemA[kc & 1][m * 17];
#pragma unroll
      for (int d = 0; d < 8; ++d)
        a.u[d] = row[kb + d + ((d >= 4) ? 4 : 0)];
    }

    // ---- B fragments: pre-swizzled, 2x b128 per lane, coalesced ----
    Frag b0, b1, b2, b3;
    {
      size_t u0 = (((size_t)(ntbase + 0) * nKC + kc) * 32 + lane) * 2;
      size_t u1 = (((size_t)(ntbase + 1) * nKC + kc) * 32 + lane) * 2;
      size_t u2 = (((size_t)(ntbase + 2) * nKC + kc) * 32 + lane) * 2;
      size_t u3 = (((size_t)(ntbase + 3) * nKC + kc) * 32 + lane) * 2;
      b0.q[0] = Wq[u0]; b0.q[1] = Wq[u0 + 1];
      b1.q[0] = Wq[u1]; b1.q[1] = Wq[u1 + 1];
      b2.q[0] = Wq[u2]; b2.q[1] = Wq[u2 + 1];
      b3.q[0] = Wq[u3]; b3.q[1] = Wq[u3 + 1];
    }

    acc0 = __builtin_amdgcn_wmma_f32_16x16x32_bf16(false, a.v, false, b0.v,
                                                   (short)0, acc0, false, false);
    acc1 = __builtin_amdgcn_wmma_f32_16x16x32_bf16(false, a.v, false, b1.v,
                                                   (short)0, acc1, false, false);
    acc2 = __builtin_amdgcn_wmma_f32_16x16x32_bf16(false, a.v, false, b2.v,
                                                   (short)0, acc2, false, false);
    acc3 = __builtin_amdgcn_wmma_f32_16x16x32_bf16(false, a.v, false, b3.v,
                                                   (short)0, acc3, false, false);
  }

  // ---- store D + bias (C/D 16x16 layout: lane = N, vgpr v = M row) ----
  const int n_local = lane & 15;
  const int mbase   = mtile + mw * 16 + (laneHi ? 8 : 0);
#pragma unroll
  for (int j = 0; j < 4; ++j) {
    const int nr = ncol0 + nw * 64 + j * 16 + n_local;
    int gcol; float bias;
    if (!ogate) { gcol = p * G3H + nr; bias = bg[p * G3H + nr]; }
    else        { gcol = NGATES + nr;  bias = bo[nr]; }
    v8f acc = (j == 0) ? acc0 : (j == 1) ? acc1 : (j == 2) ? acc2 : acc3;
#pragma unroll
    for (int v = 0; v < 8; ++v)
      gout[(size_t)(mbase + v) * NTOT + gcol] = acc[v] + bias;
  }
}

// ---------------------------------------------------------------------------
// Elementwise cell update: c_new = sig(f)*c + sig(i)*tanh(g) per part,
// h = sig(o) * tanh(sum_p c_new). c updated in place; h overwritten.
// ---------------------------------------------------------------------------
__global__ __launch_bounds__(256)
void update_kernel(const float* __restrict__ gates,  // [B, NTOT]
                   float* __restrict__ c,            // [B, P, H]
                   float* __restrict__ h)            // [B, H]
{
  int idx = blockIdx.x * 256 + threadIdx.x;          // B*H threads
  int b = idx >> 9;
  int j = idx & (H_ - 1);
  const float* g = gates + (size_t)b * NTOT;
  float s = 0.0f;
#pragma unroll
  for (int p = 0; p < P_; ++p) {
    float gi = sigmoidf_(g[p * G3H + j]);
    float gf = sigmoidf_(g[p * G3H + H_ + j]);
    float gg = tanhf(g[p * G3H + 2 * H_ + j]);
    size_t ci = ((size_t)b * P_ + p) * H_ + j;
    float cn = gf * c[ci] + gi * gg;
    c[ci] = cn;
    s += cn;
  }
  float o = sigmoidf_(g[NGATES + j]);
  h[(size_t)b * H_ + j] = o * tanhf(s);
}

// ---------------------------------------------------------------------------
// Final FC [B,H]@[H,C] + bias + row-wise log_softmax. Tiny; one block/row.
// ---------------------------------------------------------------------------
__global__ __launch_bounds__(64)
void fc_kernel(const float* __restrict__ h,
               const float* __restrict__ Wfc,   // [H, C]
               const float* __restrict__ bfc,   // [C]
               float* __restrict__ out)         // [B, C]
{
  __shared__ float hrow[H_];
  __shared__ float red[64];
  const int b = blockIdx.x, tid = threadIdx.x;
  for (int i = tid; i < H_; i += 64) hrow[i] = h[(size_t)b * H_ + i];
  __syncthreads();

  float logit = -INFINITY;
  if (tid < C_) {
    float a = bfc[tid];
    for (int k = 0; k < H_; ++k) a = fmaf(hrow[k], Wfc[(size_t)k * C_ + tid], a);
    logit = a;
  }
  red[tid] = logit; __syncthreads();
  for (int s = 32; s > 0; s >>= 1) {
    if (tid < s) red[tid] = fmaxf(red[tid], red[tid + s]);
    __syncthreads();
  }
  float m = red[0]; __syncthreads();
  float e = (tid < C_) ? expf(logit - m) : 0.0f;
  red[tid] = e; __syncthreads();
  for (int s = 32; s > 0; s >>= 1) {
    if (tid < s) red[tid] += red[tid + s];
    __syncthreads();
  }
  float lse = logf(red[0]);
  if (tid < C_) out[(size_t)b * C_ + tid] = logit - m - lse;
}

// ---------------------------------------------------------------------------
extern "C" void kernel_launch(void* const* d_in, const int* in_sizes, int n_in,
                              void* d_out, int out_size, void* d_ws, size_t ws_size,
                              hipStream_t stream) {
  (void)in_sizes; (void)n_in; (void)out_size; (void)ws_size;

  const float* x    = (const float*)d_in[0];   // [P,B,T,Fp]
  const float* Wx1  = (const float*)d_in[1];   // [P,Fp,3H]
  const float* Wh1  = (const float*)d_in[2];   // [P,H,3H]
  const float* b1   = (const float*)d_in[3];   // [P,3H]
  const float* Wxo1 = (const float*)d_in[4];   // [P*Fp,H]
  const float* Who1 = (const float*)d_in[5];   // [H,H]
  const float* bo1  = (const float*)d_in[6];   // [H]
  const float* Wx2  = (const float*)d_in[7];
  const float* Wh2  = (const float*)d_in[8];
  const float* b2   = (const float*)d_in[9];
  const float* Wxo2 = (const float*)d_in[10];
  const float* Who2 = (const float*)d_in[11];
  const float* bo2  = (const float*)d_in[12];
  const float* Wfc  = (const float*)d_in[13];  // [H,C]
  const float* bfc  = (const float*)d_in[14];  // [C]

  uint8_t* ws = (uint8_t*)d_ws;
  size_t off = 0;
  auto take = [&](size_t bytes) -> uint8_t* {
    uint8_t* r = ws + off;
    off = (off + bytes + 255) & ~(size_t)255;
    return r;
  };
  const size_t GW = 20ull * 96 * 512;          // halves per part (640x1536)
  const size_t OW = 32ull * 32 * 512;          // halves (1024x512)
  uint16_t* Wg1 = (uint16_t*)take(4 * GW * 2);
  uint16_t* Wo1 = (uint16_t*)take(OW * 2);
  uint16_t* Wg2 = (uint16_t*)take(4 * GW * 2);
  uint16_t* Wo2 = (uint16_t*)take(OW * 2);
  float* hbuf   = (float*)take((size_t)B_ * H_ * 4);
  float* c1     = (float*)take((size_t)B_ * P_ * H_ * 4);
  float* c2     = (float*)take((size_t)B_ * P_ * H_ * 4);
  float* gates  = (float*)take((size_t)B_ * NTOT * 4);

  hipMemsetAsync(hbuf, 0, (size_t)B_ * H_ * 4, stream);
  hipMemsetAsync(c1,   0, (size_t)B_ * P_ * H_ * 4, stream);
  hipMemsetAsync(c2,   0, (size_t)B_ * P_ * H_ * 4, stream);

  // Pre-swizzle weights to bf16 fragment-major layout (L2-resident, ~17 MB).
  for (int p = 0; p < P_; ++p) {
    convert_w<<<3840, 256, 0, stream>>>(Wx1 + (size_t)p * FP_ * G3H, FP_,
                                        Wh1 + (size_t)p * H_ * G3H, G3H, 20,
                                        Wg1 + (size_t)p * GW);
    convert_w<<<3840, 256, 0, stream>>>(Wx2 + (size_t)p * FP_ * G3H, FP_,
                                        Wh2 + (size_t)p * H_ * G3H, G3H, 20,
                                        Wg2 + (size_t)p * GW);
  }
  convert_w<<<2048, 256, 0, stream>>>(Wxo1, P_ * FP_, Who1, H_, 32, Wo1);
  convert_w<<<2048, 256, 0, stream>>>(Wxo2, P_ * FP_, Who2, H_, 32, Wo2);

  dim3 ggrid(52, 4);
  for (int t = 0; t < T_; ++t) {
    gates_kernel<<<ggrid, 256, 0, stream>>>(x, t, hbuf, Wg1, Wo1, b1, bo1, gates);
    update_kernel<<<(B_ * H_) / 256, 256, 0, stream>>>(gates, c1, hbuf);
    gates_kernel<<<ggrid, 256, 0, stream>>>(x, t, hbuf, Wg2, Wo2, b2, bo2, gates);
    update_kernel<<<(B_ * H_) / 256, 256, 0, stream>>>(gates, c2, hbuf);
  }
  fc_kernel<<<B_, 64, 0, stream>>>(hbuf, Wfc, bfc, (float*)d_out);
}
